// EdgeUpdaterWrapper_v01_20985210208291
// MI455X (gfx1250) — matrix operand order repaired and checked by reading, not compiled, not run on hardware
//
#include <hip/hip_runtime.h>
#include <hip/hip_bf16.h>
#include <math.h>

// ---------------- types ----------------
typedef __bf16  v16bf __attribute__((ext_vector_type(16)));
typedef __bf16  v8bf  __attribute__((ext_vector_type(8)));
typedef float   v8f   __attribute__((ext_vector_type(8)));

// dims
#define NN   512     // tokens
#define ED   128     // edge dim
#define TD   256     // token dim
#define HD   128     // down dim
#define MD   384     // mlp dim
#define ROWS 64      // edge rows per workgroup
#define KT12 12      // MD/32 k-steps
#define NT_MD 24     // MD/16 n-tiles
#define NT_ED 8      // ED/16 n-tiles
#define LNEPS 1e-5f

// workspace layout (bytes)
#define XD_OFF  0u
#define XD_SZ   (512u*128u*2u)                   // 131072
#define W1_OFF  (XD_OFF + XD_SZ)                 // 131072
#define W1_SZ   (12u*24u*32u*16u*2u)             // 294912
#define W2_OFF  (W1_OFF + W1_SZ)                 // 425984
#define W2_SZ   W1_SZ
#define WO_OFF  (W2_OFF + W2_SZ)                 // 720896
#define WO_SZ   (12u*8u*32u*16u*2u)              // 98304

// ---------------------------------------------------------------------------
// Stage 1: token layernorm + down-projection  -> x_down (bf16) [512][128]
// ---------------------------------------------------------------------------
__global__ __launch_bounds__(128)
void xdown_kernel(const float* __restrict__ h,
                  const float* __restrict__ g, const float* __restrict__ b,
                  const float* __restrict__ Wd, const float* __restrict__ bd,
                  __bf16* __restrict__ xd)
{
    __shared__ float sh[TD];
    __shared__ float red[128];
    const int i = blockIdx.x;
    const int t = threadIdx.x;

    sh[t]       = h[i * TD + t];
    sh[t + 128] = h[i * TD + 128 + t];
    __syncthreads();

    red[t] = sh[t] + sh[t + 128];
    __syncthreads();
    for (int s = 64; s > 0; s >>= 1) { if (t < s) red[t] += red[t + s]; __syncthreads(); }
    const float mu = red[0] * (1.0f / TD);
    __syncthreads();

    float d0 = sh[t] - mu, d1 = sh[t + 128] - mu;
    red[t] = d0 * d0 + d1 * d1;
    __syncthreads();
    for (int s = 64; s > 0; s >>= 1) { if (t < s) red[t] += red[t + s]; __syncthreads(); }
    const float rs = rsqrtf(red[0] * (1.0f / TD) + LNEPS);
    __syncthreads();

    sh[t]       = d0 * rs * g[t]       + b[t];
    sh[t + 128] = d1 * rs * g[t + 128] + b[t + 128];
    __syncthreads();

    float acc = bd[t];
    #pragma unroll 8
    for (int k = 0; k < TD; ++k) acc += sh[k] * Wd[(size_t)k * HD + t];
    xd[(size_t)i * HD + t] = (__bf16)acc;
}

// ---------------------------------------------------------------------------
// Stage 2: pack a f32 [K x N] weight into bf16 WMMA-B tile layout.
// Tile (kt,nt): lane l -> column n = nt*16+(l&15), K half = (l>=16)?16:0,
// 16 contiguous halfs; stored at out[((kt*ntilesN+nt)*32 + lane)*16].
// ---------------------------------------------------------------------------
__global__ __launch_bounds__(256)
void pack_kernel(const float* __restrict__ W, int Ndim,
                 __bf16* __restrict__ out, int ntilesN, int total)
{
    const int id = blockIdx.x * 256 + threadIdx.x;
    if (id >= total) return;
    const int lane = id & 31;
    const int tile = id >> 5;
    const int kt = tile / ntilesN;
    const int nt = tile - kt * ntilesN;
    const int n  = nt * 16 + (lane & 15);
    const int kb = kt * 32 + ((lane >> 4) << 4);
    __bf16* o = out + ((size_t)tile * 32 + lane) * 16;
    #pragma unroll
    for (int t = 0; t < 16; ++t)
        o[t] = (__bf16)W[(size_t)(kb + t) * Ndim + n];
}

// ---------------------------------------------------------------------------
// Fragment loaders
// ---------------------------------------------------------------------------
__device__ __forceinline__ v16bf cat8(v8bf lo, v8bf hi) {
    return __builtin_shufflevector(lo, hi, 0,1,2,3,4,5,6,7,8,9,10,11,12,13,14,15);
}

// A-fragment from LDS, row-major bf16 with stride MD.
__device__ __forceinline__ v16bf load_a(const __bf16* sA, int mt, int kt, int lane) {
    const int row = mt * 16 + (lane & 15);
    const __bf16* base = sA + row * MD + kt * 32 + ((lane >> 4) << 4);
    v8bf lo = *(const v8bf*)(base);
    v8bf hi = *(const v8bf*)(base + 8);
    return cat8(lo, hi);
}

// B-fragment from the packed weight stream: 32 bytes per lane, contiguous.
__device__ __forceinline__ v16bf load_b(const __bf16* pW, int kt, int nt, int ntilesN, int lane) {
    const __bf16* bp = pW + ((size_t)((kt * ntilesN + nt) * 32) + lane) * 16;
    v8bf lo = *(const v8bf*)(bp);
    v8bf hi = *(const v8bf*)(bp + 8);
    return cat8(lo, hi);
}

#define WMMA_BF16(A, B, C) \
    __builtin_amdgcn_wmma_f32_16x16x32_bf16(false, (A), false, (B), (short)0, (C), false, false)

// 2x2 register-blocked GEMM inner product over K = MD (12 k-steps).
// One A pair (rows mt0,mt0+1) is reused by two B columns: 4 WMMAs per
// 2 ds_load_b128-pairs + 2 global_load_b128-pairs.
__device__ __forceinline__ void gemm_2x2(const __bf16* sA, const __bf16* pW,
                                         int mt0, int nt0, int ntilesN, int lane,
                                         float bias0, float bias1,
                                         v8f& acc00, v8f& acc01,
                                         v8f& acc10, v8f& acc11)
{
    #pragma unroll
    for (int v = 0; v < 8; ++v) { acc00[v] = bias0; acc10[v] = bias0; }
    #pragma unroll
    for (int v = 0; v < 8; ++v) { acc01[v] = bias1; acc11[v] = bias1; }
    #pragma unroll
    for (int kt = 0; kt < KT12; ++kt) {
        v16bf a0 = load_a(sA, mt0,     kt, lane);
        v16bf a1 = load_a(sA, mt0 + 1, kt, lane);
        v16bf b0 = load_b(pW, kt, nt0,     ntilesN, lane);
        v16bf b1 = load_b(pW, kt, nt0 + 1, ntilesN, lane);
        acc00 = WMMA_BF16(a0, b0, acc00);
        acc01 = WMMA_BF16(a0, b1, acc01);
        acc10 = WMMA_BF16(a1, b0, acc10);
        acc11 = WMMA_BF16(a1, b1, acc11);
    }
}

// C/D f32 layout: lane -> col n = lane&15; VGPR v -> row v + 8*(lane>=16).
__device__ __forceinline__ void store_relu_bf16(__bf16* dst, int stride,
                                                int mt, int nt, int lane, const v8f& acc) {
    const int n  = nt * 16 + (lane & 15);
    const int mb = mt * 16 + ((lane >> 4) << 3);
    #pragma unroll
    for (int v = 0; v < 8; ++v)
        dst[(mb + v) * stride + n] = (__bf16)fmaxf(acc[v], 0.f);
}

__device__ __forceinline__ void store_relu_residual(__bf16* dst, int stride,
                                                    int mt, int nt, int lane, const v8f& acc) {
    const int n  = nt * 16 + (lane & 15);
    const int mb = mt * 16 + ((lane >> 4) << 3);
    #pragma unroll
    for (int v = 0; v < 8; ++v) {
        const int idx = (mb + v) * stride + n;
        float x = fmaxf(acc[v], 0.f) + (float)dst[idx];
        dst[idx] = (__bf16)x;
    }
}

__device__ __forceinline__ void store_f32(float* dst, int stride,
                                          int mt, int nt, int lane, const v8f& acc) {
    const int n  = nt * 16 + (lane & 15);
    const int mb = mt * 16 + ((lane >> 4) << 3);
    #pragma unroll
    for (int v = 0; v < 8; ++v)
        dst[(mb + v) * stride + n] = acc[v];
}

// ---------------------------------------------------------------------------
// Main fused edge kernel. 256 threads (8 waves), 64 edge rows per block.
// Wave w -> (mh = w>>2 : rows mh*32..+31, ng = w&3 : N column group).
// ---------------------------------------------------------------------------
__global__ __launch_bounds__(256)
void edge_kernel(const float* __restrict__ p,
                 const float* __restrict__ ge, const float* __restrict__ be,
                 const float* __restrict__ b1, const float* __restrict__ b2,
                 const float* __restrict__ bo,
                 const float* __restrict__ go, const float* __restrict__ bbo,
                 const __bf16* __restrict__ xd,
                 const __bf16* __restrict__ pW1,
                 const __bf16* __restrict__ pW2,
                 const __bf16* __restrict__ pWo,
                 float* __restrict__ out)
{
    extern __shared__ char smem[];
    __bf16* s_zin  = (__bf16*)(smem);                 // 64*384 bf16 = 49152 B
    __bf16* s_h1   = (__bf16*)(smem + 49152);         // 64*384 bf16 = 49152 B
    float*  s_pln  = (float*)(smem + 98304);          // 64*128 f32  = 32768 B
    float*  s_zout = (float*)(smem + 131072);         // 64*128 f32  = 32768 B
    float*  s_mu   = (float*)(smem + 163840);         // 64 f32
    float*  s_rs   = s_mu + 64;                       // 64 f32

    const int tid  = threadIdx.x;
    const int lane = tid & 31;
    const int wave = tid >> 5;
    const int i    = blockIdx.x >> 3;
    const int j0   = (blockIdx.x & 7) * ROWS;

    // ---- phase 0: load p tile (coalesced, contiguous 32KB) ----
    const float* pbase = p + ((size_t)i * NN + j0) * ED;
    for (int idx = tid; idx < ROWS * ED; idx += 256) s_pln[idx] = pbase[idx];
    __syncthreads();

    // ---- phase 1: per-row LN statistics for p ----
    if (tid < ROWS) {
        const float* r = s_pln + tid * ED;
        float s = 0.f, ss = 0.f;
        #pragma unroll 8
        for (int c = 0; c < ED; ++c) { float v = r[c]; s += v; ss += v * v; }
        float mu = s * (1.0f / ED);
        float var = ss * (1.0f / ED) - mu * mu;
        s_mu[tid] = mu;
        s_rs[tid] = rsqrtf(var + LNEPS);
    }
    __syncthreads();

    // ---- phase 2: p_ln (keep f32 for residual) + build bf16 z_in rows ----
    for (int idx = tid; idx < ROWS * ED; idx += 256) {
        const int r = idx >> 7, c = idx & 127;
        float v = (s_pln[idx] - s_mu[r]) * s_rs[r] * ge[c] + be[c];
        s_pln[idx] = v;
        s_zin[r * MD + 2 * HD + c] = (__bf16)v;
        s_zin[r * MD + c]          = xd[(size_t)i * HD + c];
        s_zin[r * MD + HD + c]     = xd[(size_t)(j0 + r) * HD + c];
    }
    __syncthreads();

    const int mh  = wave >> 2;        // 0..1 : which 32 rows
    const int ng  = wave & 3;         // 0..3 : N column group
    const int mt0 = mh * 2;           // first of two 16-row M tiles

    v8f a00, a01, a10, a11;

    // ---- GEMM1: h1 = relu(z_in @ W1 + b1) ----
    #pragma unroll 1
    for (int q = 0; q < 3; ++q) {
        const int nt0 = ng * 6 + q * 2;
        const float bias0 = b1[nt0 * 16 + (lane & 15)];
        const float bias1 = b1[(nt0 + 1) * 16 + (lane & 15)];
        gemm_2x2(s_zin, pW1, mt0, nt0, NT_MD, lane, bias0, bias1, a00, a01, a10, a11);
        store_relu_bf16(s_h1, MD, mt0,     nt0,     lane, a00);
        store_relu_bf16(s_h1, MD, mt0,     nt0 + 1, lane, a01);
        store_relu_bf16(s_h1, MD, mt0 + 1, nt0,     lane, a10);
        store_relu_bf16(s_h1, MD, mt0 + 1, nt0 + 1, lane, a11);
    }
    __syncthreads();

    // ---- GEMM2: a3 = relu(h1 @ W2 + b2) + z_in  (written back into s_zin) ----
    #pragma unroll 1
    for (int q = 0; q < 3; ++q) {
        const int nt0 = ng * 6 + q * 2;
        const float bias0 = b2[nt0 * 16 + (lane & 15)];
        const float bias1 = b2[(nt0 + 1) * 16 + (lane & 15)];
        gemm_2x2(s_h1, pW2, mt0, nt0, NT_MD, lane, bias0, bias1, a00, a01, a10, a11);
        store_relu_residual(s_zin, MD, mt0,     nt0,     lane, a00);
        store_relu_residual(s_zin, MD, mt0,     nt0 + 1, lane, a01);
        store_relu_residual(s_zin, MD, mt0 + 1, nt0,     lane, a10);
        store_relu_residual(s_zin, MD, mt0 + 1, nt0 + 1, lane, a11);
    }
    __syncthreads();

    // ---- GEMM3: z = a3 @ Wo + bo  -> s_zout (f32) ----
    {
        const int nt0 = ng * 2;
        const float bias0 = bo[nt0 * 16 + (lane & 15)];
        const float bias1 = bo[(nt0 + 1) * 16 + (lane & 15)];
        gemm_2x2(s_zin, pWo, mt0, nt0, NT_ED, lane, bias0, bias1, a00, a01, a10, a11);
        store_f32(s_zout, ED, mt0,     nt0,     lane, a00);
        store_f32(s_zout, ED, mt0,     nt0 + 1, lane, a01);
        store_f32(s_zout, ED, mt0 + 1, nt0,     lane, a10);
        store_f32(s_zout, ED, mt0 + 1, nt0 + 1, lane, a11);
    }
    __syncthreads();

    // ---- output layernorm stats ----
    if (tid < ROWS) {
        const float* r = s_zout + tid * ED;
        float s = 0.f, ss = 0.f;
        #pragma unroll 8
        for (int c = 0; c < ED; ++c) { float v = r[c]; s += v; ss += v * v; }
        float mu = s * (1.0f / ED);
        float var = ss * (1.0f / ED) - mu * mu;
        s_mu[tid] = mu;
        s_rs[tid] = rsqrtf(var + LNEPS);
    }
    __syncthreads();

    // ---- out = p_ln + LN(z)  (coalesced f32 store) ----
    float* obase = out + ((size_t)i * NN + j0) * ED;
    for (int idx = tid; idx < ROWS * ED; idx += 256) {
        const int r = idx >> 7, c = idx & 127;
        float z = (s_zout[idx] - s_mu[r]) * s_rs[r] * go[c] + bbo[c];
        obase[idx] = s_pln[idx] + z;
    }
}

extern "C" void kernel_launch(void* const* d_in, const int* in_sizes, int n_in,
                              void* d_out, int out_size, void* d_ws, size_t ws_size,
                              hipStream_t stream) {
    const float* p     = (const float*)d_in[0];
    const float* h     = (const float*)d_in[1];
    const float* ge    = (const float*)d_in[2];
    const float* be    = (const float*)d_in[3];
    const float* gt    = (const float*)d_in[4];
    const float* bt    = (const float*)d_in[5];
    const float* Wd    = (const float*)d_in[6];
    const float* bd    = (const float*)d_in[7];
    const float* W1    = (const float*)d_in[8];
    const float* b1    = (const float*)d_in[9];
    const float* W2    = (const float*)d_in[10];
    const float* b2    = (const float*)d_in[11];
    const float* Wo    = (const float*)d_in[12];
    const float* bo    = (const float*)d_in[13];
    const float* go    = (const float*)d_in[14];
    const float* bbo   = (const float*)d_in[15];
    float* out = (float*)d_out;

    char* ws = (char*)d_ws;
    __bf16* xd  = (__bf16*)(ws + XD_OFF);
    __bf16* pW1 = (__bf16*)(ws + W1_OFF);
    __bf16* pW2 = (__bf16*)(ws + W2_OFF);
    __bf16* pWo = (__bf16*)(ws + WO_OFF);

    // Stage 1: token LN + down projection
    xdown_kernel<<<NN, 128, 0, stream>>>(h, gt, bt, Wd, bd, xd);

    // Stage 2: pack weights into WMMA B-tile bf16 layout
    pack_kernel<<<36, 256, 0, stream>>>(W1, MD, pW1, NT_MD, 12 * 24 * 32);
    pack_kernel<<<36, 256, 0, stream>>>(W2, MD, pW2, NT_MD, 12 * 24 * 32);
    pack_kernel<<<12, 256, 0, stream>>>(Wo, ED, pWo, NT_ED, 12 * 8 * 32);

    // Stage 3: fused edge MLP; 4096 blocks x 256 threads, 160.5 KB dynamic LDS
    const size_t lds_bytes = 163840 + 512;
    edge_kernel<<<NN * (NN / ROWS), 256, lds_bytes, stream>>>(
        p, ge, be, b1, b2, bo, go, bbo, xd, pW1, pW2, pWo, out);
}
